// MultiHeadSelfAttention_16776142258417
// MI455X (gfx1250) — matrix operand order repaired and checked by reading, not compiled, not run on hardware
//
#include <hip/hip_runtime.h>
#include <hip/hip_bf16.h>

typedef __bf16 bf16;
typedef bf16  v16bf __attribute__((ext_vector_type(16)));
typedef bf16  v8bf  __attribute__((ext_vector_type(8)));
typedef bf16  v4bf  __attribute__((ext_vector_type(4)));
typedef float v8f   __attribute__((ext_vector_type(8)));
typedef unsigned int u32x4 __attribute__((ext_vector_type(4)));
typedef int i32x4 __attribute__((ext_vector_type(4)));
typedef int i32x8 __attribute__((ext_vector_type(8)));

#define WMMA_BF16(a, b, c) \
  __builtin_amdgcn_wmma_f32_16x16x32_bf16(false, (a), false, (b), (short)0, (c), false, false)

// ---------------------------------------------------------------------------
// Fragment loaders matching CDNA5 16-bit WMMA VGPR layouts (ISA 7.12.2).
// A 16x32: lane = row (lane&15); lanes<16 hold K {0..7,16..23}, lanes>=16 hold
//          K {8..15,24..31}  -> two b128 loads at +0/+16 (elements).
// B 32x16: lane = col (lane&15); lanes<16 hold K {0..7,16..23}, lanes>=16 hold
//          K {8..15,24..31}  -> two b128 loads at +0/+16 (elements).
// ---------------------------------------------------------------------------
__device__ __forceinline__ v16bf frag_a(const bf16* base, int lane, int stride) {
  const bf16* p = base + (lane & 15) * stride + ((lane & 16) ? 8 : 0);
  v8bf lo = *(const v8bf*)(p);
  v8bf hi = *(const v8bf*)(p + 16);
  return __builtin_shufflevector(lo, hi, 0,1,2,3,4,5,6,7,8,9,10,11,12,13,14,15);
}
__device__ __forceinline__ v16bf frag_b(const bf16* base, int lane, int stride) {
  const bf16* p = base + (lane & 15) * stride + ((lane & 16) ? 8 : 0);
  v8bf lo = *(const v8bf*)(p);
  v8bf hi = *(const v8bf*)(p + 16);
  return __builtin_shufflevector(lo, hi, 0,1,2,3,4,5,6,7,8,9,10,11,12,13,14,15);
}

// ---------------------------------------------------------------------------
// Tensor Data Mover: 2D bf16 tile load Global -> LDS (ISA ch.8 D# layout).
// group0: [1:0]=count=1, [63:32]=lds_addr, [120:64]=global_addr, [127:126]=2
// group1: [17:16]=data_size(1=2B), [79:48]=tensor_dim0, [111:80]=tensor_dim1,
//         [127:112]=tile_dim0, [143:128]=tile_dim1, [207:160]=dim0_stride
// Arity differs across toolchains (probe-verified): clang-22 = 5 args,
// clang-23 (therock) = 6 args.
// ---------------------------------------------------------------------------
__device__ __forceinline__ void tdm_load_2d_bf16(const void* gaddr, unsigned lds_off,
                                                 unsigned dim0, unsigned dim1,
                                                 unsigned stride0,
                                                 unsigned tile0, unsigned tile1) {
  unsigned long long ga = (unsigned long long)gaddr;
  u32x4 g0;
  g0[0] = 1u;                                               // count=1 (valid)
  g0[1] = lds_off;                                          // lds_addr (bytes)
  g0[2] = (unsigned)(ga & 0xFFFFFFFFu);                     // global_addr lo
  g0[3] = (unsigned)((ga >> 32) & 0x01FFFFFFu) | (2u << 30);// addr hi | type=2
  i32x8 g1;
  g1[0] = (int)(1u << 16);                                  // data_size = 2B
  g1[1] = (int)((dim0 & 0xFFFFu) << 16);                    // tensor_dim0 lo
  g1[2] = (int)((dim0 >> 16) | ((dim1 & 0xFFFFu) << 16));   // dim0 hi | dim1 lo
  g1[3] = (int)((dim1 >> 16) | (tile0 << 16));              // dim1 hi | tile0
  g1[4] = (int)tile1;                                       // tile_dim1
  g1[5] = (int)stride0;                                     // dim0_stride lo
  g1[6] = 0;
  g1[7] = 0;
  i32x4 z4 = {0, 0, 0, 0};
#if defined(__clang_major__) && (__clang_major__ >= 23)
  i32x8 z8 = {0, 0, 0, 0, 0, 0, 0, 0};
  __builtin_amdgcn_tensor_load_to_lds(g0, g1, z4, z4, z8, 0);
#else
  __builtin_amdgcn_tensor_load_to_lds(g0, g1, z4, z4, 0);
#endif
}

// ---------------------------------------------------------------------------
// f32 -> bf16 conversion (vectorized 4-wide)
// ---------------------------------------------------------------------------
__global__ __launch_bounds__(256) void cvt_f32_bf16(const float* __restrict__ in,
                                                    bf16* __restrict__ out, int n4) {
  int i = blockIdx.x * 256 + threadIdx.x;
  if (i < n4) {
    float4 v = ((const float4*)in)[i];
    v4bf o;
    o[0] = (bf16)v.x; o[1] = (bf16)v.y; o[2] = (bf16)v.z; o[3] = (bf16)v.w;
    ((v4bf*)out)[i] = o;
  }
}

// ---------------------------------------------------------------------------
// bf16 GEMM: C(f32, MxN) = A(bf16, MxK row-major) * B(bf16, KxN row-major)
// Block tile 128x128, BK=32, 8 waves (4 M x 2 N), each wave 32x64 via
// 2x4 v_wmma_f32_16x16x32_bf16. A tile staged by the Tensor Data Mover
// (one TDM descriptor per k-step, issued by wave 0, TENSORcnt-waited);
// B tile staged transposed by all threads. N,K compile-time for cheap
// immediate-offset addressing.
// ---------------------------------------------------------------------------
template<int N, int K>
__global__ __launch_bounds__(256) void gemm_bf16(const bf16* __restrict__ A,
                                                 const bf16* __restrict__ B,
                                                 float* __restrict__ C,
                                                 int M) {
  __shared__ bf16 As[128 * 32];   // [row][k]   <- TDM destination
  __shared__ bf16 Bs[128 * 32];   // [n][k]     <- transposed staging
  const int tid  = threadIdx.x;
  const int wave = tid >> 5;
  const int lane = tid & 31;
  const int bm = blockIdx.x * 128;
  const int bn = blockIdx.y * 128;
  const int wm = (wave >> 1) * 32;
  const int wn = (wave & 1) * 64;
  const unsigned as_off = (unsigned)(size_t)(void*)As;  // LDS byte offset

  v8f acc[2][4] = {};

  for (int k0 = 0; k0 < K; k0 += 32) {
    // --- A tile (128x32) via Tensor Data Mover, wave 0 only
    if (wave == 0) {
      tdm_load_2d_bf16(A + (size_t)bm * K + k0, as_off,
                       (unsigned)K, (unsigned)M, (unsigned)K, 32u, 128u);
      __builtin_amdgcn_s_wait_tensorcnt(0);
    }
    // --- B tile (32x128) transposed into Bs[n][k]
#pragma unroll
    for (int i = 0; i < 2; i++) {
      int idx = tid + i * 256;
      int r = idx >> 4;          // k within tile
      int c = (idx & 15) * 8;    // n start
      v8bf t = *(const v8bf*)(B + (size_t)(k0 + r) * N + bn + c);
#pragma unroll
      for (int j = 0; j < 8; j++) Bs[(c + j) * 32 + r] = t[j];
    }
    __syncthreads();

    v16bf af[2], bfr[4];
#pragma unroll
    for (int i = 0; i < 2; i++) af[i] = frag_a(As + (wm + i * 16) * 32, lane, 32);
#pragma unroll
    for (int j = 0; j < 4; j++) bfr[j] = frag_b(Bs + (wn + j * 16) * 32, lane, 32);
#pragma unroll
    for (int i = 0; i < 2; i++)
#pragma unroll
      for (int j = 0; j < 4; j++) acc[i][j] = WMMA_BF16(af[i], bfr[j], acc[i][j]);
    __syncthreads();
  }

  // C layout: VGPR r -> (M=r lanes 0-15, M=r+8 lanes 16-31), N = lane&15.
  // Single base pointer; all offsets compile-time -> immediate-offset stores.
  const int col    = lane & 15;
  const int rowoff = (lane & 16) ? 8 : 0;
  float* Cp = C + (size_t)(bm + wm + rowoff) * N + bn + wn + col;
#pragma unroll
  for (int i = 0; i < 2; i++)
#pragma unroll
    for (int j = 0; j < 4; j++)
#pragma unroll
      for (int r = 0; r < 8; r++)
        Cp[(size_t)(i * 16 + r) * N + j * 16] = acc[i][j][r];
}

// ---------------------------------------------------------------------------
// RoPE + head-split + V transpose.
// qkv f32 [B,S,3,H,hd] -> Q,K bf16 [B,H,S,hd] (1/sqrt(hd) folded into Q),
//                         Vt bf16 [B,H,hd,S]
// ---------------------------------------------------------------------------
__global__ __launch_bounds__(256) void rope_split(const float* __restrict__ qkv,
                                                  const long long* __restrict__ pos,
                                                  bf16* __restrict__ Qb,
                                                  bf16* __restrict__ Kb,
                                                  bf16* __restrict__ Vt) {
  const int D = 2048, S = 2048, H = 16, hd = 128;
  int idx = blockIdx.x * 256 + threadIdx.x;
  int j = idx & 63;
  int h = (idx >> 6) & 15;
  int s = (idx >> 10) & 2047;
  int b = idx >> 21;

  size_t base = ((size_t)(b * S + s) * 3) * D + h * hd;
  float q1 = qkv[base + 2 * j],         q2 = qkv[base + 2 * j + 1];
  float k1 = qkv[base + D + 2 * j],     k2 = qkv[base + D + 2 * j + 1];
  float v1 = qkv[base + 2 * D + 2 * j], v2 = qkv[base + 2 * D + 2 * j + 1];

  float p    = (float)pos[s];
  float freq = __powf(10000.0f, -(float)(2 * j) * (1.0f / 128.0f));
  float ang  = p * freq;
  float sn, cs;
  __sincosf(ang, &sn, &cs);
  const float scale = 0.08838834764831845f;  // 1/sqrt(128)

  size_t hb = ((size_t)(b * H + h) * S + s) * hd;
  Qb[hb + 2 * j]     = (bf16)((q1 * cs - q2 * sn) * scale);
  Qb[hb + 2 * j + 1] = (bf16)((q1 * sn + q2 * cs) * scale);
  Kb[hb + 2 * j]     = (bf16)(k1 * cs - k2 * sn);
  Kb[hb + 2 * j + 1] = (bf16)(k1 * sn + k2 * cs);

  size_t vb = ((size_t)(b * H + h) * hd) * S;
  Vt[vb + (size_t)(2 * j) * S + s]     = (bf16)v1;
  Vt[vb + (size_t)(2 * j + 1) * S + s] = (bf16)v2;
}

// ---------------------------------------------------------------------------
// Causal flash attention. Grid: (S/64, B*H). 128 threads = 4 waves.
// Each wave owns 16 query rows: Q frags in regs, O = 16x128 f32 in regs,
// online softmax with width-16 shuffle butterflies; P round-trips through a
// private per-wave LDS slab (in-order DS pipe, no barriers).
// ---------------------------------------------------------------------------
__global__ __launch_bounds__(128) void flash_attn(const bf16* __restrict__ Qb,
                                                  const bf16* __restrict__ Kb,
                                                  const bf16* __restrict__ Vt,
                                                  bf16* __restrict__ Ob) {
  const int S = 2048, hd = 128;
  __shared__ bf16 Plds[4][16 * 32];

  const int wave  = threadIdx.x >> 5;
  const int lane  = threadIdx.x & 31;
  const int bh    = blockIdx.y;               // b*16 + h
  const int qbase = blockIdx.x * 64 + wave * 16;

  const bf16* Qp = Qb + ((size_t)bh * S + qbase) * hd;
  const bf16* Kp = Kb + (size_t)bh * S * hd;
  const bf16* Vp = Vt + (size_t)bh * hd * S;

  v16bf qf[4];
#pragma unroll
  for (int ks = 0; ks < 4; ks++) qf[ks] = frag_a(Qp + ks * 32, lane, hd);

  v8f o[8] = {};
  float mrow[8], lrow[8];
#pragma unroll
  for (int r = 0; r < 8; r++) { mrow[r] = -3.0e38f; lrow[r] = 0.0f; }

  const int col    = lane & 15;
  const int rowoff = (lane & 16) ? 8 : 0;
  const int kend   = qbase + 16;              // keys needed: [0, qbase+15]

  for (int kb = 0; kb < kend; kb += 32) {
    if (kb + 32 < kend) {
      __builtin_prefetch(Kp + (size_t)(kb + 32) * hd, 0, 1);
      __builtin_prefetch(Vp + kb + 32, 0, 1);
    }
    // ---- score tiles: rows = q, cols = keys kb..kb+15 / kb+16..kb+31
    v8f s0 = {}, s1 = {};
#pragma unroll
    for (int ks = 0; ks < 4; ks++) {
      v16bf b0 = frag_b(Kp + (size_t)kb * hd + ks * 32, lane, hd);
      v16bf b1 = frag_b(Kp + (size_t)(kb + 16) * hd + ks * 32, lane, hd);
      s0 = WMMA_BF16(qf[ks], b0, s0);
      s1 = WMMA_BF16(qf[ks], b1, s1);
    }
    // ---- causal mask + online softmax
    float scl[8];
#pragma unroll
    for (int r = 0; r < 8; r++) {
      int m   = qbase + rowoff + r;
      float a0 = ((kb + col)      <= m) ? s0[r] : -3.0e38f;
      float a1 = ((kb + 16 + col) <= m) ? s1[r] : -3.0e38f;
      float mx = fmaxf(a0, a1);
      mx = fmaxf(mx, __shfl_xor(mx, 1, 16));
      mx = fmaxf(mx, __shfl_xor(mx, 2, 16));
      mx = fmaxf(mx, __shfl_xor(mx, 4, 16));
      mx = fmaxf(mx, __shfl_xor(mx, 8, 16));
      float mnew = fmaxf(mrow[r], mx);
      float sc   = __expf(mrow[r] - mnew);
      mrow[r] = mnew;
      scl[r]  = sc;
      float p0 = __expf(a0 - mnew);
      float p1 = __expf(a1 - mnew);
      float sum = p0 + p1;
      sum += __shfl_xor(sum, 1, 16);
      sum += __shfl_xor(sum, 2, 16);
      sum += __shfl_xor(sum, 4, 16);
      sum += __shfl_xor(sum, 8, 16);
      lrow[r] = lrow[r] * sc + sum;
      Plds[wave][(rowoff + r) * 32 + col]      = (bf16)p0;
      Plds[wave][(rowoff + r) * 32 + 16 + col] = (bf16)p1;
    }
    // ---- rescale running O, then accumulate P @ V
#pragma unroll
    for (int t = 0; t < 8; t++)
#pragma unroll
      for (int r = 0; r < 8; r++) o[t][r] *= scl[r];

    v16bf pf = frag_a(&Plds[wave][0], lane, 32);
#pragma unroll
    for (int t = 0; t < 8; t++) {
      v16bf vf = frag_b(Vp + (size_t)(t * 16) * S + kb, lane, S);
      o[t] = WMMA_BF16(pf, vf, o[t]);
    }
  }

  // ---- normalize and write [B,S,D] bf16
  const int b = bh >> 4, h = bh & 15;
  bf16* Op = Ob + ((size_t)b * S + qbase) * 2048 + h * 128;
#pragma unroll
  for (int t = 0; t < 8; t++)
#pragma unroll
    for (int r = 0; r < 8; r++) {
      float val = o[t][r] / lrow[r];
      Op[(size_t)(rowoff + r) * 2048 + t * 16 + col] = (bf16)val;
    }
}

// ---------------------------------------------------------------------------
// Host launch
// ---------------------------------------------------------------------------
extern "C" void kernel_launch(void* const* d_in, const int* in_sizes, int n_in,
                              void* d_out, int out_size, void* d_ws, size_t ws_size,
                              hipStream_t stream) {
  const int B = 2, S = 2048, D = 2048, H = 16, hd = 128;
  const int M = B * S;            // 4096
  const int N1 = 3 * D;           // 6144

  const float*     x    = (const float*)d_in[0];
  const long long* pos  = (const long long*)d_in[1];
  const float*     wqkv = (const float*)d_in[2];
  const float*     wo   = (const float*)d_in[3];
  float*           out  = (float*)d_out;
  char*            ws   = (char*)d_ws;

  // workspace layout (bytes)
  size_t oXB  = 0;                                  // x bf16      [M,D]
  size_t oWQ  = oXB  + (size_t)M * D * 2;           // wqkv bf16   [D,3D]
  size_t oWO  = oWQ  + (size_t)D * N1 * 2;          // wo bf16     [D,D]
  size_t oQKV = oWO  + (size_t)D * D * 2;           // qkv f32     [M,3D]
  size_t oQ   = oQKV + (size_t)M * N1 * 4;          // Q bf16      [B,H,S,hd]
  size_t oK   = oQ   + (size_t)B * H * S * hd * 2;  // K bf16      [B,H,S,hd]
  size_t oV   = oK   + (size_t)B * H * S * hd * 2;  // Vt bf16     [B,H,hd,S]
  size_t oA   = oV   + (size_t)B * H * S * hd * 2;  // attn bf16   [B,S,D]

  bf16* xb    = (bf16*)(ws + oXB);
  bf16* wqkvb = (bf16*)(ws + oWQ);
  bf16* wob   = (bf16*)(ws + oWO);
  float* qkv  = (float*)(ws + oQKV);
  bf16* Qb    = (bf16*)(ws + oQ);
  bf16* Kb    = (bf16*)(ws + oK);
  bf16* Vt    = (bf16*)(ws + oV);
  bf16* attnb = (bf16*)(ws + oA);

  // 1) downcast inputs to bf16
  {
    int n4 = (M * D) / 4;
    cvt_f32_bf16<<<(n4 + 255) / 256, 256, 0, stream>>>(x, xb, n4);
  }
  {
    int n4 = (D * N1) / 4;
    cvt_f32_bf16<<<(n4 + 255) / 256, 256, 0, stream>>>(wqkv, wqkvb, n4);
  }
  {
    int n4 = (D * D) / 4;
    cvt_f32_bf16<<<(n4 + 255) / 256, 256, 0, stream>>>(wo, wob, n4);
  }

  // 2) qkv = x @ w_qkv   (4096 x 6144 x 2048)
  gemm_bf16<6144, 2048><<<dim3(M / 128, N1 / 128), 256, 0, stream>>>(xb, wqkvb, qkv, M);

  // 3) RoPE + head split + V transpose
  {
    int total = B * S * H * (hd / 2);   // 4,194,304
    rope_split<<<total / 256, 256, 0, stream>>>(qkv, pos, Qb, Kb, Vt);
  }

  // 4) causal flash attention
  flash_attn<<<dim3(S / 64, B * H), 128, 0, stream>>>(Qb, Kb, Vt, attnb);

  // 5) out = attn @ w_o   (4096 x 2048 x 2048), f32 straight to d_out
  gemm_bf16<2048, 2048><<<dim3(M / 128, D / 128), 256, 0, stream>>>(attnb, wob, out, M);
}
